// MinkEngDeConvRelu_68676527063292
// MI455X (gfx1250) — matrix operand order repaired and checked by reading, not compiled, not run on hardware
//
#include <hip/hip_runtime.h>

typedef __attribute__((ext_vector_type(16))) __bf16 v16bf;
typedef __attribute__((ext_vector_type(8)))  float  v8f;

#define K_OFF   27
#define M_TOT   100000
#define C_IN    256
#define C_OUT   128
#define N_OUTR  200000
#define M_TILE  64
#define TILES_M ((M_TOT + M_TILE - 1) / M_TILE)   // 1563

struct AB32 { uint4 lo, hi; };   // 32 bytes = one 16-element bf16 fragment

__device__ __forceinline__ unsigned pk_bf16(float a, float b) {
  // round-to-nearest-even f32 -> bf16, packed pair
  unsigned ua = __float_as_uint(a), ub = __float_as_uint(b);
  ua = (ua + 0x7fffu + ((ua >> 16) & 1u)) >> 16;
  ub = (ub + 0x7fffu + ((ub >> 16) & 1u)) >> 16;
  return ua | (ub << 16);
}

// ---------------------------------------------------------------- zero out
__global__ __launch_bounds__(256) void zero_out_kernel(float4* __restrict__ out) {
  out[(size_t)blockIdx.x * 256 + threadIdx.x] = make_float4(0.f, 0.f, 0.f, 0.f);
}

// ------------------------------------------- weight f32 -> bf16 B-swizzle
// Layout: tile (k, kb, nb) holds the 32x16 B fragment in per-lane register order, contiguous k (same order as A):
//   slot = lane*16 + e ; K = kb*32 + (lane>=16 ? 16 : 0) + e ; N = nb*16 + (lane&15)
// so each lane's fragment is 32 contiguous bytes (two b128 loads).
__global__ __launch_bounds__(256) void prep_w_kernel(const float* __restrict__ W,
                                                     unsigned short* __restrict__ Wb) {
  int tid  = blockIdx.x * 256 + threadIdx.x;        // exactly 27*8*8*512 threads
  int e    = tid & 15;
  int lane = (tid >> 4) & 31;
  int nb   = (tid >> 9) & 7;
  int kb   = (tid >> 12) & 7;
  int k    = tid >> 15;
  int K    = kb * 32 + ((lane >> 4) << 4) + e;
  int N    = nb * 16 + (lane & 15);
  float v  = W[((size_t)k * C_IN + K) * C_OUT + N];
  unsigned u = __float_as_uint(v);
  Wb[tid] = (unsigned short)((u + 0x7fffu + ((u >> 16) & 1u)) >> 16);
}

// -------------------------------------------------- gather-GEMM-scatter
__global__ __launch_bounds__(256) void gemm_scatter_kernel(
    const float* __restrict__ feats, const unsigned short* __restrict__ Wb,
    const int* __restrict__ in_map, const int* __restrict__ out_map,
    float* __restrict__ out)
{
  __shared__ __align__(16) unsigned short As[M_TILE][C_IN];   // 32 KB bf16 A tile
  __shared__ int imap_s[M_TILE];
  __shared__ int omap_s[M_TILE];

  const int tid = threadIdx.x;
  const int bid = blockIdx.x;
  const int k   = bid / TILES_M;
  const int m0  = (bid % TILES_M) * M_TILE;

  if (tid < M_TILE) {
    int m  = m0 + tid;
    int mc = (m < M_TOT) ? m : (M_TOT - 1);
    imap_s[tid] = in_map[(size_t)k * M_TOT + mc];
    omap_s[tid] = (m < M_TOT) ? out_map[(size_t)k * M_TOT + m] : -1;
  }
  __syncthreads();

  // Gather 64 rows of feats (f32) -> LDS bf16. 4096 float4 chunks, 16/thread.
  // Consecutive threads hit consecutive 16B chunks of the same row: coalesced
  // global loads, conflict-free 8B LDS stores. feats (102 MB) is L2-resident.
  #pragma unroll
  for (int it = 0; it < 16; ++it) {
    int c    = it * 256 + tid;          // 0..4095
    int r    = c >> 6;                  // row 0..63
    int coff = (c & 63) << 2;           // col 0..252 step 4
    const float4 v = *(const float4*)(feats + (size_t)imap_s[r] * C_IN + coff);
    *(uint2*)&As[r][coff] = make_uint2(pk_bf16(v.x, v.y), pk_bf16(v.z, v.w));
  }
  __syncthreads();

  const int lane = tid & 31;
  const int w    = tid >> 5;            // wave id == N-tile index (0..7)
  const int hi16 = lane >> 4;           // 0 | 1 (lane half)
  const int mrow = lane & 15;

  const v8f vzero = {0.f, 0.f, 0.f, 0.f, 0.f, 0.f, 0.f, 0.f};
  v8f acc[4] = {vzero, vzero, vzero, vzero};

  #pragma unroll
  for (int kb = 0; kb < 8; ++kb) {
    // B fragment: pre-swizzled, contiguous 32B per lane (two b128 in a clause)
    const uint4* bp = (const uint4*)Wb
                      + ((size_t)((k * 8 + kb) * 8 + w) * 64) + lane * 2;
    AB32 bfr;
    bfr.lo = bp[0];
    bfr.hi = bp[1];

    // Issue ALL four A-fragment loads first so a single dscnt wait covers the
    // whole clause and the 4 WMMAs can run back-to-back.
    const int kb0 = kb * 32 + hi16 * 16; // contiguous A layout: chunks at K and K+8
    AB32 afr[4];
    #pragma unroll
    for (int i = 0; i < 4; ++i) {
      const int row = i * 16 + mrow;
      afr[i].lo = *(const uint4*)&As[row][kb0];
      afr[i].hi = *(const uint4*)&As[row][kb0 + 8];
    }

    const v16bf bmat = __builtin_bit_cast(v16bf, bfr);
    #pragma unroll
    for (int i = 0; i < 4; ++i) {
      acc[i] = __builtin_amdgcn_wmma_f32_16x16x32_bf16(
                   false, __builtin_bit_cast(v16bf, afr[i]), false, bmat,
                   (short)0, acc[i], false, false);
    }
  }

  // Scatter-add per C/D layout: VGPR v holds M=v (lanes 0-15) / M=v+8 (16-31)
  const int col = w * 16 + mrow;
  #pragma unroll
  for (int i = 0; i < 4; ++i) {
    #pragma unroll
    for (int v = 0; v < 8; ++v) {
      int ml = i * 16 + v + (hi16 << 3);
      int om = omap_s[ml];
      if (om >= 0)
        __hip_atomic_fetch_add(out + (size_t)om * C_OUT + col, acc[i][v],
                               __ATOMIC_RELAXED, __HIP_MEMORY_SCOPE_AGENT);
    }
  }
}

// ---------------------------------------------------------- bias + ReLU
__global__ __launch_bounds__(256) void bias_relu_kernel(float4* __restrict__ out,
                                                        const float4* __restrict__ bias4) {
  size_t i = (size_t)blockIdx.x * 256 + threadIdx.x;   // float4 index
  float4 b = bias4[i & 31];                            // 128 cols / 4
  float4 v = out[i];
  v.x = fmaxf(v.x + b.x, 0.f);
  v.y = fmaxf(v.y + b.y, 0.f);
  v.z = fmaxf(v.z + b.z, 0.f);
  v.w = fmaxf(v.w + b.w, 0.f);
  out[i] = v;
}

extern "C" void kernel_launch(void* const* d_in, const int* in_sizes, int n_in,
                              void* d_out, int out_size, void* d_ws, size_t ws_size,
                              hipStream_t stream) {
  const float* feats   = (const float*)d_in[0];   // [100000, 256] f32
  const float* weight  = (const float*)d_in[1];   // [27, 256, 128] f32
  const float* bias    = (const float*)d_in[2];   // [128] f32
  const int*   in_map  = (const int*)d_in[3];     // [27, 100000] i32
  const int*   out_map = (const int*)d_in[4];     // [27, 100000] i32
  float*       out     = (float*)d_out;           // [200000, 128] f32
  unsigned short* Wb   = (unsigned short*)d_ws;   // 1.77 MB bf16 swizzled weights

  // 200000*128 floats = 6,400,000 float4 = 25000 blocks of 256
  zero_out_kernel<<<25000, 256, 0, stream>>>((float4*)out);
  // 27*256*128 = 884,736 elements = 3456 blocks of 256
  prep_w_kernel<<<3456, 256, 0, stream>>>(weight, Wb);
  // 27 offsets * 1563 row-tiles
  gemm_scatter_kernel<<<K_OFF * TILES_M, 256, 0, stream>>>(feats, Wb, in_map, out_map, out);
  bias_relu_kernel<<<25000, 256, 0, stream>>>((float4*)out, (const float4*)bias);
}
